// MoDAAttention_7267084664934
// MI455X (gfx1250) — compile-verified, matrix operand order checked by reading, not run on hardware
//
#include <hip/hip_runtime.h>
#include <hip/hip_bf16.h>
#include <math.h>

// ---------------- problem constants (match reference) ----------------
#define T_SEQ   2048
#define DMODEL  2048
#define HQ      16
#define HK      8
#define HD      128
#define NL      4
#define GQA     (HQ / HK)
#define SCALE   0.08838834764831845f   // 128^-0.5

typedef __bf16 bf16;
typedef __attribute__((ext_vector_type(16))) __bf16 v16bf;
typedef __attribute__((ext_vector_type(8)))  __bf16 v8bf;
typedef __attribute__((ext_vector_type(8)))  float  v8f;
typedef __attribute__((ext_vector_type(4)))  unsigned int u32x4;
typedef __attribute__((ext_vector_type(8)))  unsigned int u32x8;

// =====================================================================
// Tensor Data Mover: 2-D tile load Global -> LDS (2-group D#)
// =====================================================================
__device__ __forceinline__ void tdm_load_2d(unsigned int lds_addr,
                                            const void* gptr,
                                            unsigned int tile_w,     // elements
                                            unsigned int tile_h,     // rows
                                            unsigned int row_stride, // elements
                                            unsigned int pad_en,
                                            unsigned int pad_interval_code,
                                            unsigned int pad_amount_code) {
    unsigned long long ga = (unsigned long long)(size_t)gptr;
    u32x4 g0;
    g0[0] = 1u;                                            // count = 1
    g0[1] = lds_addr;
    g0[2] = (unsigned int)ga;
    g0[3] = (unsigned int)((ga >> 32) & 0x1ffffffu) | (2u << 30); // type=2
    u32x8 g1;
    g1[0] = (1u << 16) | (pad_en << 20) |
            (pad_interval_code << 22) | (pad_amount_code << 25);
    g1[1] = (tile_w & 0xffffu) << 16;                      // tensor_dim0 lo
    g1[2] = (tile_w >> 16) | ((tile_h & 0xffffu) << 16);   // dim0 hi | dim1 lo
    g1[3] = (tile_h >> 16) | ((tile_w & 0xffffu) << 16);   // dim1 hi | tile_dim0
    g1[4] = (tile_h & 0xffffu);                            // tile_dim1, tile_dim2=0
    g1[5] = row_stride;                                    // dim0_stride lo
    g1[6] = 0u;
    g1[7] = 0u;
    asm volatile("tensor_load_to_lds %0, %1" :: "s"(g0), "s"(g1) : "memory");
}

__device__ __forceinline__ void tdm_wait() {
    __builtin_amdgcn_s_wait_tensorcnt(0);
}

// =====================================================================
// Kernel 1a: fp32 -> bf16 convert (grid-stride)
// =====================================================================
__global__ void f32_to_bf16_kernel(const float* __restrict__ src,
                                   bf16* __restrict__ dst, int n) {
    for (int i = blockIdx.x * blockDim.x + threadIdx.x; i < n;
         i += gridDim.x * blockDim.x) {
        dst[i] = (bf16)src[i];
    }
}

// =====================================================================
// Kernel 1b: fp32 [R][Ccols] -> bf16 transposed [Ccols][R] (LDS 32x32 tiles)
// =====================================================================
__global__ __launch_bounds__(256)
void transpose_to_bf16_kernel(const float* __restrict__ src,
                              bf16* __restrict__ dst, int R, int Ccols) {
    __shared__ float t[32][33];
    int c0 = blockIdx.x * 32, r0 = blockIdx.y * 32;
    int tx = threadIdx.x & 31, ty = threadIdx.x >> 5;
#pragma unroll
    for (int i = ty; i < 32; i += 8)
        t[i][tx] = src[(size_t)(r0 + i) * Ccols + c0 + tx];
    __syncthreads();
#pragma unroll
    for (int i = ty; i < 32; i += 8)
        dst[(size_t)(c0 + i) * R + r0 + tx] = (bf16)t[tx][i];
}

// =====================================================================
// WMMA fragment helper (ISA 7.12.2, wave32)
// Row fragment: lane -> row = lane&15; hl = lane>>4;
//   elements 0..7  = row[hl*8 .. +8), elements 8..15 = row[16+hl*8 .. +8)
// C 16x16 f32: lane -> N = lane&15; vgpr r -> M = r + 8*hl
// =====================================================================
__device__ __forceinline__ v16bf load_frag_row(const bf16* rowbase, int hl) {
    v8bf lo = *reinterpret_cast<const v8bf*>(rowbase + hl * 8);
    v8bf hi = *reinterpret_cast<const v8bf*>(rowbase + 16 + hl * 8);
    return __builtin_shufflevector(lo, hi, 0, 1, 2, 3, 4, 5, 6, 7,
                                   8, 9, 10, 11, 12, 13, 14, 15);
}

// =====================================================================
// Kernel 2: bf16 GEMM  C[M,N](f32) = A[M,K] * Bt[N,K]^T  (Bt is N-major)
// 128 threads = 4 waves (2x2), block tile 128x128, wave tile 64x64,
// K-step 32, TDM double-buffered staging with hardware row padding.
// =====================================================================
#define GB_M 128
#define GB_N 128
#define GB_K 32
#define LS_STRIDE 40   // 32 el + 4 DW pad (80B rows, 16B aligned)

template <int M, int N, int K>
__global__ __launch_bounds__(128)
void gemm_bf16_kernel(const bf16* __restrict__ A, const bf16* __restrict__ Bt,
                      float* __restrict__ C) {
    __shared__ bf16 As[2][GB_M * LS_STRIDE];
    __shared__ bf16 Bs[2][GB_N * LS_STRIDE];

    const int tid  = threadIdx.x;
    const int lane = tid & 31;
    const int wave = tid >> 5;
    const int wm = wave >> 1;
    const int wn = wave & 1;
    const int m0 = blockIdx.y * GB_M;
    const int n0 = blockIdx.x * GB_N;
    const int hl = lane >> 4;

    v8f c[4][4];
#pragma unroll
    for (int i = 0; i < 4; ++i)
#pragma unroll
        for (int j = 0; j < 4; ++j) c[i][j] = (v8f)(0.0f);

    const int kTiles = K / GB_K;

    if (wave == 0) {
        tdm_load_2d((unsigned int)(size_t)&As[0][0], &A[(size_t)m0 * K],
                    GB_K, GB_M, (unsigned int)K, 1u, 3u, 3u);
        tdm_load_2d((unsigned int)(size_t)&Bs[0][0], &Bt[(size_t)n0 * K],
                    GB_K, GB_N, (unsigned int)K, 1u, 3u, 3u);
        tdm_wait();
    }
    __syncthreads();

    int cur = 0;
    for (int kt = 0; kt < kTiles; ++kt) {
        const int nxt = cur ^ 1;
        if (wave == 0 && kt + 1 < kTiles) {
            const int kb = (kt + 1) * GB_K;
            tdm_load_2d((unsigned int)(size_t)&As[nxt][0],
                        &A[(size_t)m0 * K + kb], GB_K, GB_M,
                        (unsigned int)K, 1u, 3u, 3u);
            tdm_load_2d((unsigned int)(size_t)&Bs[nxt][0],
                        &Bt[(size_t)n0 * K + kb], GB_K, GB_N,
                        (unsigned int)K, 1u, 3u, 3u);
        }

        v16bf af[4], bfr[4];
#pragma unroll
        for (int i = 0; i < 4; ++i) {
            int m = wm * 64 + i * 16 + (lane & 15);
            af[i] = load_frag_row(&As[cur][m * LS_STRIDE], hl);
        }
#pragma unroll
        for (int j = 0; j < 4; ++j) {
            int n = wn * 64 + j * 16 + (lane & 15);
            bfr[j] = load_frag_row(&Bs[cur][n * LS_STRIDE], hl);
        }

#pragma unroll
        for (int i = 0; i < 4; ++i)
#pragma unroll
            for (int j = 0; j < 4; ++j)
                c[i][j] = __builtin_amdgcn_wmma_f32_16x16x32_bf16(
                    false, af[i], false, bfr[j], (short)0, c[i][j], false, false);

        if (wave == 0 && kt + 1 < kTiles) tdm_wait();
        __syncthreads();
        cur = nxt;
    }

    {
        float* cbase = &C[(size_t)(m0 + wm * 64 + 8 * hl) * N +
                          n0 + wn * 64 + (lane & 15)];
#pragma unroll
        for (int i = 0; i < 4; ++i)
#pragma unroll
            for (int j = 0; j < 4; ++j)
#pragma unroll
                for (int r = 0; r < 8; ++r)
                    cbase[(size_t)(i * 16 + r) * N + j * 16] = c[i][j][r];
    }
}

// =====================================================================
// Kernel 3: RoPE (optional) + repack, with output pre-scaling
//   layout 0: [T,H,HD] f32 -> [H][T][HD] bf16
//   layout 1: [T,H,HD] f32 -> [H][HD][T] bf16 (dim-major, for V)
// =====================================================================
__global__ void rope_pack_kernel(const float* __restrict__ src,
                                 const float* __restrict__ cosT,
                                 const float* __restrict__ sinT,
                                 bf16* __restrict__ dst, int H, int applyRope,
                                 int dimMajor, float outScale) {
    int idx = blockIdx.x * blockDim.x + threadIdx.x;
    int d = idx & (HD - 1);
    int h = (idx / HD) % H;
    int t = idx / (HD * H);
    float v = src[(size_t)t * H * HD + h * HD + d];
    if (applyRope) {
        float c = cosT[t * HD + d];
        float s = sinT[t * HD + d];
        float other = (d < HD / 2)
            ? -src[(size_t)t * H * HD + h * HD + d + HD / 2]
            :  src[(size_t)t * H * HD + h * HD + d - HD / 2];
        v = v * c + other * s;
    }
    v *= outScale;
    if (dimMajor)
        dst[((size_t)h * HD + d) * T_SEQ + t] = (bf16)v;
    else
        dst[((size_t)h * T_SEQ + t) * HD + d] = (bf16)v;
}

// =====================================================================
// Kernel 4: flash attention with joint depth+sequence softmax
// grid: (T/64, HQ), block: 128 threads = 4 waves, 16 query rows per wave
// Q is pre-scaled by 1/sqrt(HD). K (key-major) and V (dim-major) tiles
// staged by TDM, double-buffered to overlap DMA with WMMA.
// =====================================================================
#define QTILE 64
#define KTILE 64
#define PS_STRIDE 72
#define VS_STRIDE 72   // 64 el + 4 DW pad

__global__ __launch_bounds__(128)
void attn_kernel(const bf16* __restrict__ Qb,   // [HQ][T][HD], pre-scaled
                 const bf16* __restrict__ Kb,   // [HK][T][HD]
                 const bf16* __restrict__ Vt,   // [HK][HD][T]
                 const float* __restrict__ depth_k,  // [L][HK][T][HD]
                 const float* __restrict__ depth_v,  // [L][HK][T][HD]
                 bf16* __restrict__ Ob) {       // [T][HQ*HD]
    __shared__ bf16  Ks[2][KTILE * HD];         // key-major 64x128
    __shared__ bf16  Vts[2][HD * VS_STRIDE];    // dim-major 128x(64+8)
    __shared__ bf16  Ps[4][16 * PS_STRIDE];
    __shared__ float dls[QTILE][NL];

    const int h  = blockIdx.y;
    const int kh = h / GQA;
    const int qt = blockIdx.x;
    const int q0 = qt * QTILE;
    const int tid  = threadIdx.x;
    const int lane = tid & 31;
    const int wave = tid >> 5;
    const int hl   = lane >> 4;
    const int qrow0 = q0 + wave * 16;

    // kick off DMA of key tile 0 immediately; overlaps the prologue work
    if (wave == 0) {
        tdm_load_2d((unsigned int)(size_t)&Ks[0][0],
                    &Kb[((size_t)kh * T_SEQ) * HD], HD, KTILE, HD, 0u, 0u, 0u);
        tdm_load_2d((unsigned int)(size_t)&Vts[0][0],
                    &Vt[(size_t)kh * HD * T_SEQ], KTILE, HD, T_SEQ, 1u, 4u, 3u);
    }

    // ---- preload Q fragments (A-matrix), 4 k-slabs of HD=128
    v16bf qf[4];
    {
        const bf16* qrow = &Qb[((size_t)h * T_SEQ + qrow0 + (lane & 15)) * HD];
#pragma unroll
        for (int ks = 0; ks < 4; ++ks) qf[ks] = load_frag_row(qrow + ks * 32, hl);
    }

    // ---- depth logits: 64 rows x 4 slots (Q pre-scaled -> no extra scale)
#pragma unroll
    for (int p = tid; p < QTILE * NL; p += 128) {
        int row = p >> 2, l = p & 3;
        int q = q0 + row;
        const bf16*   qp = &Qb[((size_t)h * T_SEQ + q) * HD];
        const float4* kp = reinterpret_cast<const float4*>(
            &depth_k[(((size_t)l * HK + kh) * T_SEQ + q) * HD]);
        float acc = 0.f;
#pragma unroll 4
        for (int d4 = 0; d4 < HD / 4; ++d4) {
            float4 kv = kp[d4];
            acc += (float)qp[4 * d4 + 0] * kv.x + (float)qp[4 * d4 + 1] * kv.y +
                   (float)qp[4 * d4 + 2] * kv.z + (float)qp[4 * d4 + 3] * kv.w;
        }
        dls[row][l] = acc;
    }
    __syncthreads();

    // ---- init running softmax state from depth slots
    float mrow[8], srow[8];
    v8f acc[8];
#pragma unroll
    for (int r = 0; r < 8; ++r) {
        int row = wave * 16 + r + 8 * hl;
        float m = dls[row][0];
#pragma unroll
        for (int l = 1; l < NL; ++l) m = fmaxf(m, dls[row][l]);
        float s = 0.f;
#pragma unroll
        for (int l = 0; l < NL; ++l) s += __expf(dls[row][l] - m);
        mrow[r] = m;
        srow[r] = s;
    }
#pragma unroll
    for (int f = 0; f < 8; ++f) {
#pragma unroll
        for (int r = 0; r < 8; ++r) {
            int row = wave * 16 + r + 8 * hl;
            int q = q0 + row;
            int n = f * 16 + (lane & 15);
            float a = 0.f;
#pragma unroll
            for (int l = 0; l < NL; ++l)
                a += __expf(dls[row][l] - mrow[r]) *
                     depth_v[(((size_t)l * HK + kh) * T_SEQ + q) * HD + n];
            acc[f][r] = a;
        }
    }

    if (wave == 0) tdm_wait();
    __syncthreads();   // tile 0 resident

    // ---- stream causal key tiles (double-buffered TDM)
    int cur = 0;
    for (int kt = 0; kt <= qt; ++kt) {
        const int k0 = kt * KTILE;
        const int nxt = cur ^ 1;
        if (wave == 0 && kt < qt) {
            const int k1 = (kt + 1) * KTILE;
            tdm_load_2d((unsigned int)(size_t)&Ks[nxt][0],
                        &Kb[((size_t)kh * T_SEQ + k1) * HD],
                        HD, KTILE, HD, 0u, 0u, 0u);
            tdm_load_2d((unsigned int)(size_t)&Vts[nxt][0],
                        &Vt[(size_t)kh * HD * T_SEQ + k1],
                        KTILE, HD, T_SEQ, 1u, 4u, 3u);
        }

        // S = Q * K^T : 4 key-frags x 4 dim-slabs (row frags from Ks)
        v8f sf[4];
#pragma unroll
        for (int j = 0; j < 4; ++j) sf[j] = (v8f)(0.0f);
#pragma unroll
        for (int j = 0; j < 4; ++j) {
            const bf16* krow = &Ks[cur][(j * 16 + (lane & 15)) * HD];
#pragma unroll
            for (int ks = 0; ks < 4; ++ks) {
                v16bf bkf = load_frag_row(krow + ks * 32, hl);
                sf[j] = __builtin_amdgcn_wmma_f32_16x16x32_bf16(
                    false, qf[ks], false, bkf, (short)0, sf[j], false, false);
            }
        }

        // causal mask (Q pre-scaled, no extra multiply)
        float p[4][8];
#pragma unroll
        for (int j = 0; j < 4; ++j)
#pragma unroll
            for (int r = 0; r < 8; ++r) {
                int key = k0 + j * 16 + (lane & 15);
                int q   = q0 + wave * 16 + r + 8 * hl;
                p[j][r] = (key > q) ? -__builtin_inff() : sf[j][r];
            }

        // row max / exp / row sum (16-lane xor reductions), update state
        float alpha[8];
#pragma unroll
        for (int r = 0; r < 8; ++r) {
            float rm = p[0][r];
#pragma unroll
            for (int j = 1; j < 4; ++j) rm = fmaxf(rm, p[j][r]);
#pragma unroll
            for (int off = 1; off < 16; off <<= 1)
                rm = fmaxf(rm, __shfl_xor(rm, off, 32));
            float newm = fmaxf(mrow[r], rm);
            alpha[r] = __expf(mrow[r] - newm);
            float rs = 0.f;
#pragma unroll
            for (int j = 0; j < 4; ++j) {
                p[j][r] = __expf(p[j][r] - newm);
                rs += p[j][r];
            }
#pragma unroll
            for (int off = 1; off < 16; off <<= 1)
                rs += __shfl_xor(rs, off, 32);
            srow[r] = srow[r] * alpha[r] + rs;
            mrow[r] = newm;
        }
#pragma unroll
        for (int f = 0; f < 8; ++f)
#pragma unroll
            for (int r = 0; r < 8; ++r) acc[f][r] *= alpha[r];

        // stage P (C-layout -> LDS -> A-layout)
#pragma unroll
        for (int j = 0; j < 4; ++j)
#pragma unroll
            for (int r = 0; r < 8; ++r)
                Ps[wave][(r + 8 * hl) * PS_STRIDE + j * 16 + (lane & 15)] =
                    (bf16)p[j][r];
        __syncthreads();

        // acc += P * V : 2 key-slabs x 8 dim-frags (row frags from Vts)
#pragma unroll
        for (int kv = 0; kv < 2; ++kv) {
            v16bf ap = load_frag_row(
                &Ps[wave][(lane & 15) * PS_STRIDE + kv * 32], hl);
#pragma unroll
            for (int f = 0; f < 8; ++f) {
                v16bf bv = load_frag_row(
                    &Vts[cur][(f * 16 + (lane & 15)) * VS_STRIDE + kv * 32], hl);
                acc[f] = __builtin_amdgcn_wmma_f32_16x16x32_bf16(
                    false, ap, false, bv, (short)0, acc[f], false, false);
            }
        }

        if (wave == 0 && kt < qt) tdm_wait();
        __syncthreads();
        cur = nxt;
    }

    // ---- finalize: out = acc / srow
#pragma unroll
    for (int f = 0; f < 8; ++f)
#pragma unroll
        for (int r = 0; r < 8; ++r) {
            int row = wave * 16 + r + 8 * hl;
            int q = q0 + row;
            int n = f * 16 + (lane & 15);
            Ob[(size_t)q * (HQ * HD) + h * HD + n] = (bf16)(acc[f][r] / srow[r]);
        }
}

// =====================================================================
// host-side launcher
// =====================================================================
extern "C" void kernel_launch(void* const* d_in, const int* in_sizes, int n_in,
                              void* d_out, int out_size, void* d_ws, size_t ws_size,
                              hipStream_t stream) {
    const float* x   = (const float*)d_in[0];
    const float* dk  = (const float*)d_in[1];
    const float* dv  = (const float*)d_in[2];
    const float* cosT = (const float*)d_in[3];
    const float* sinT = (const float*)d_in[4];
    const float* Wq  = (const float*)d_in[5];
    const float* Wk  = (const float*)d_in[6];
    const float* Wv  = (const float*)d_in[7];
    const float* Wo  = (const float*)d_in[8];
    float* out = (float*)d_out;

    size_t off = 0;
    auto carve = [&](size_t bytes) {
        void* p = (char*)d_ws + off;
        off += (bytes + 255) & ~(size_t)255;
        return p;
    };
    const size_t NQ = (size_t)T_SEQ * HQ * HD;
    const size_t NK = (size_t)T_SEQ * HK * HD;
    const size_t NX = (size_t)T_SEQ * DMODEL;

    bf16*  xb   = (bf16*)carve(NX * 2);
    bf16*  WqT  = (bf16*)carve(NX * 2);
    bf16*  WkT  = (bf16*)carve((size_t)DMODEL * HK * HD * 2);
    bf16*  WvT  = (bf16*)carve((size_t)DMODEL * HK * HD * 2);
    bf16*  WoT  = (bf16*)carve(NX * 2);
    float* Qf   = (float*)carve(NQ * 4);
    float* Kf   = (float*)carve(NK * 4);
    float* Vf   = (float*)carve(NK * 4);
    bf16*  Qb   = (bf16*)carve(NQ * 2);
    bf16*  Kb   = (bf16*)carve(NK * 2);
    bf16*  Vt   = (bf16*)carve(NK * 2);
    bf16*  Ob   = (bf16*)carve(NQ * 2);

    f32_to_bf16_kernel<<<2048, 256, 0, stream>>>(x, xb, (int)NX);
    transpose_to_bf16_kernel<<<dim3((HQ * HD) / 32, DMODEL / 32), 256, 0, stream>>>(
        Wq, WqT, DMODEL, HQ * HD);
    transpose_to_bf16_kernel<<<dim3((HK * HD) / 32, DMODEL / 32), 256, 0, stream>>>(
        Wk, WkT, DMODEL, HK * HD);
    transpose_to_bf16_kernel<<<dim3((HK * HD) / 32, DMODEL / 32), 256, 0, stream>>>(
        Wv, WvT, DMODEL, HK * HD);
    transpose_to_bf16_kernel<<<dim3(DMODEL / 32, (HQ * HD) / 32), 256, 0, stream>>>(
        Wo, WoT, HQ * HD, DMODEL);

    gemm_bf16_kernel<T_SEQ, HQ * HD, DMODEL>
        <<<dim3((HQ * HD) / GB_N, T_SEQ / GB_M), 128, 0, stream>>>(xb, WqT, Qf);
    gemm_bf16_kernel<T_SEQ, HK * HD, DMODEL>
        <<<dim3((HK * HD) / GB_N, T_SEQ / GB_M), 128, 0, stream>>>(xb, WkT, Kf);
    gemm_bf16_kernel<T_SEQ, HK * HD, DMODEL>
        <<<dim3((HK * HD) / GB_N, T_SEQ / GB_M), 128, 0, stream>>>(xb, WvT, Vf);

    // Q pre-scaled by 1/sqrt(HD); V packed dim-major for transposed PV frags
    rope_pack_kernel<<<(int)(NQ / 256), 256, 0, stream>>>(Qf, cosT, sinT, Qb, HQ, 1, 0, SCALE);
    rope_pack_kernel<<<(int)(NK / 256), 256, 0, stream>>>(Kf, cosT, sinT, Kb, HK, 1, 0, 1.0f);
    rope_pack_kernel<<<(int)(NK / 256), 256, 0, stream>>>(Vf, cosT, sinT, Vt, HK, 0, 1, 1.0f);

    attn_kernel<<<dim3(T_SEQ / QTILE, HQ), 128, 0, stream>>>(Qb, Kb, Vt, dk, dv, Ob);

    gemm_bf16_kernel<T_SEQ, DMODEL, HQ * HD>
        <<<dim3(DMODEL / GB_N, T_SEQ / GB_M), 128, 0, stream>>>(Ob, WoT, out);
}
